// MoE_13065290514703
// MI455X (gfx1250) — matrix operand order repaired and checked by reading, not compiled, not run on hardware
//
#include <hip/hip_runtime.h>
#include <stdint.h>

#define B_ 8192
#define D_ 3072
#define E_ 8
#define H_ 2048
#define O_ 512

typedef unsigned short u16;
typedef __bf16 bf16;
typedef bf16 v16bf __attribute__((ext_vector_type(16)));
typedef u16  v16u  __attribute__((ext_vector_type(16)));
typedef u16  v8u   __attribute__((ext_vector_type(8)));
typedef float v8f  __attribute__((ext_vector_type(8)));
typedef int  v4i   __attribute__((ext_vector_type(4)));

#if __has_builtin(__builtin_amdgcn_global_load_async_to_lds_b128)
#define HAVE_ASYNC_LDS 1
typedef __attribute__((address_space(1))) v4i* as1_v4i_p;
typedef __attribute__((address_space(3))) v4i* as3_v4i_p;
#else
#define HAVE_ASYNC_LDS 0
#endif

union Frag16 { v16bf bf; v16u u; };

__device__ __forceinline__ u16 f32_to_bf16(float f) {
  union { float f; uint32_t u; } c; c.f = f;
  uint32_t u = c.u;
  u += 0x7FFFu + ((u >> 16) & 1u);   // round-to-nearest-even
  return (u16)(u >> 16);
}

// Build a v16bf fragment from two contiguous 8-element (16-byte) runs.
__device__ __forceinline__ v16bf make_frag(const u16* p_lo, const u16* p_hi) {
  v8u lo = *(const v8u*)p_lo;
  v8u hi = *(const v8u*)p_hi;
  Frag16 f;
#pragma unroll
  for (int i = 0; i < 8; ++i) { f.u[i] = lo[i]; f.u[8 + i] = hi[i]; }
  return f.bf;
}

// 16-byte global -> LDS copy; async (CDNA5 ASYNCcnt path) when available.
__device__ __forceinline__ void copy16_g2l(const u16* gp, u16* lp) {
#if HAVE_ASYNC_LDS
  v4i* g = (v4i*)gp;   // drop const, reinterpret as 16-byte vector (generic)
  v4i* l = (v4i*)lp;
  __builtin_amdgcn_global_load_async_to_lds_b128(
      (as1_v4i_p)g, (as3_v4i_p)l, 0, 0);
#else
  *(v8u*)lp = *(const v8u*)gp;
#endif
}

__device__ __forceinline__ void wait_async_copies() {
#if HAVE_ASYNC_LDS
  asm volatile("s_wait_asynccnt 0x0" ::: "memory");
#endif
}

// ---------------------------------------------------------------------------
// Transpose + convert: W [E][DM][NM] f32  ->  Wt [E][NM][DM] bf16
// (makes the K dimension contiguous for WMMA B-fragments)
// ---------------------------------------------------------------------------
__global__ __launch_bounds__(256) void transpose_to_bf16(
    const float* __restrict__ W, u16* __restrict__ Wt, int DM, int NM) {
  __shared__ u16 tile[32][33];
  const int e  = blockIdx.z;
  const int d0 = blockIdx.y * 32;
  const int n0 = blockIdx.x * 32;
  const int tx = threadIdx.x, ty = threadIdx.y;  // block (32, 8)
  const float* src = W + (size_t)e * DM * NM;
#pragma unroll
  for (int r = ty; r < 32; r += 8)
    tile[r][tx] = f32_to_bf16(src[(size_t)(d0 + r) * NM + n0 + tx]);
  __syncthreads();
  u16* dst = Wt + (size_t)e * NM * DM;
#pragma unroll
  for (int r = ty; r < 32; r += 8)
    dst[(size_t)(n0 + r) * DM + d0 + tx] = tile[tx][r];
}

// ---------------------------------------------------------------------------
// Gating: logits = x @ w_gate, top-2 softmax, route rows to experts,
// accumulate importance/load, convert x -> bf16.
// ---------------------------------------------------------------------------
__global__ __launch_bounds__(256) void gate_kernel(
    const float* __restrict__ x, const float* __restrict__ w_gate,
    u16* __restrict__ xbf, int* __restrict__ counts,
    float* __restrict__ importance,
    int* __restrict__ rows_list, float* __restrict__ gate_list) {
  const int row = blockIdx.x;
  const int tid = threadIdx.x;
  __shared__ float slog[E_];
  if (tid < E_) slog[tid] = 0.f;
  __syncthreads();

  float part[E_];
#pragma unroll
  for (int e = 0; e < E_; ++e) part[e] = 0.f;
  for (int d = tid; d < D_; d += 256) {
    float xv = x[(size_t)row * D_ + d];
    xbf[(size_t)row * D_ + d] = f32_to_bf16(xv);
#pragma unroll
    for (int e = 0; e < E_; ++e) part[e] += xv * w_gate[d * E_ + e];
  }
#pragma unroll
  for (int e = 0; e < E_; ++e) atomicAdd(&slog[e], part[e]);
  __syncthreads();

  if (tid == 0) {
    int i1 = 0;
#pragma unroll
    for (int e = 1; e < E_; ++e) if (slog[e] > slog[i1]) i1 = e;  // first max = top_k tie rule
    int i2 = (i1 == 0) ? 1 : 0;
#pragma unroll
    for (int e = 0; e < E_; ++e)
      if (e != i1 && slog[e] > slog[i2]) i2 = e;
    float v1 = slog[i1], v2 = slog[i2];
    float m  = fmaxf(v1, v2);
    float g1 = __expf(v1 - m), g2 = __expf(v2 - m);
    float s  = g1 + g2;
    g1 /= s; g2 /= s;
    int p1 = atomicAdd(&counts[i1], 1);
    rows_list[(size_t)i1 * B_ + p1] = row;
    gate_list[(size_t)i1 * B_ + p1] = g1;
    int p2 = atomicAdd(&counts[i2], 1);
    rows_list[(size_t)i2 * B_ + p2] = row;
    gate_list[(size_t)i2 * B_ + p2] = g2;
    atomicAdd(&importance[i1], g1);
    atomicAdd(&importance[i2], g2);
  }
}

// ---------------------------------------------------------------------------
// Aux loss: 0.01 * (cv^2(importance) + cv^2(load)), ddof = 1.
// ---------------------------------------------------------------------------
__global__ void aux_kernel(const float* __restrict__ importance,
                           const int* __restrict__ counts,
                           float* __restrict__ out_aux) {
  float mi = 0.f, ml = 0.f;
  for (int e = 0; e < E_; ++e) { mi += importance[e]; ml += (float)counts[e]; }
  mi /= (float)E_; ml /= (float)E_;
  float vi = 0.f, vl = 0.f;
  for (int e = 0; e < E_; ++e) {
    float a = importance[e] - mi; vi += a * a;
    float b = (float)counts[e] - ml; vl += b * b;
  }
  vi /= (float)(E_ - 1); vl /= (float)(E_ - 1);
  out_aux[0] = 0.01f * (vi / (mi * mi + 1e-10f) + vl / (ml * ml + 1e-10f));
}

// ---------------------------------------------------------------------------
// Routed expert FFN, bf16 WMMA. Block = 512 threads = 16 wave32s.
// One block: 16 rows of one expert.
//   stage:  s_x[16][3072] <- gathered x rows (async global->LDS, 96 KB)
//   layer1: h[16][2048] = relu(s_x @ W1t[e]^T + b1) -> s_h (bf16, 64 KB)
//   layer2: out[16][512] = s_h @ W2t[e]^T + b2
//   combined[row] += gate * exp(out)
// Wave w owns N-cols [w*128, w*128+128) in layer1, [w*32, w*32+32) in layer2.
// ---------------------------------------------------------------------------
__global__ __launch_bounds__(512) void ffn_kernel(
    const u16* __restrict__ xbf, const u16* __restrict__ W1t,
    const float* __restrict__ b1, const u16* __restrict__ W2t,
    const float* __restrict__ b2, const int* __restrict__ counts,
    const int* __restrict__ rows_list, const float* __restrict__ gate_list,
    float* __restrict__ combined) {
  const int e    = blockIdx.y;
  const int tile = blockIdx.x;
  const int cnt  = counts[e];
  if (tile * 16 >= cnt) return;

  __shared__ u16 s_x[16 * D_];  // 96 KB gathered x tile (bf16)
  __shared__ u16 s_h[16 * H_];  // 64 KB h tile (bf16)

  const int tid  = threadIdx.x;
  const int wave = tid >> 5;
  const int lane = tid & 31;
  const int half = lane >> 4;   // 0: lanes 0-15, 1: lanes 16-31
  const int l16  = lane & 15;

  // ---- stage x tile: 16 rows x 3072 bf16 = 6144 16-byte chunks ----
  {
    const int CHUNKS_PER_ROW = D_ / 8;  // 384
#pragma unroll
    for (int i = 0; i < 12; ++i) {      // 12 * 512 = 6144
      const int c  = tid + i * 512;
      const int r  = c / CHUNKS_PER_ROW;
      const int cc = c - r * CHUNKS_PER_ROW;
      const int idx = tile * 16 + r;
      const int cl  = (idx < cnt) ? idx : (cnt - 1);
      const int grow = rows_list[(size_t)e * B_ + cl];
      copy16_g2l(xbf + (size_t)grow * D_ + cc * 8, s_x + r * D_ + cc * 8);
    }
    wait_async_copies();
    __syncthreads();
  }

  // Per-lane K offsets per the CDNA5 16-bit WMMA VGPR layouts.
  const int kA = half ? 8 : 0;    // A: runs {kA..kA+7} and {kA+16..kA+23}
  const int kB = half ? 8 : 0;    // B: runs {kB..kB+7} and {kB+16..kB+23}

  const v8f vzero = {0.f, 0.f, 0.f, 0.f, 0.f, 0.f, 0.f, 0.f};

  // ---------------- layer 1 ----------------
  const int n_base = wave * 128;  // 16 waves cover H_ = 2048
  const u16* xrow = s_x + l16 * D_;     // A source in LDS
  const u16* w1base = W1t + ((size_t)e * H_ + n_base + l16) * D_ + kB;

  v8f acc[8];
#pragma unroll
  for (int t = 0; t < 8; ++t) acc[t] = vzero;

  for (int k0 = 0; k0 < D_; k0 += 32) {
    // Load A (LDS) and ALL 8 B fragments (global) into distinct registers
    // first, so load latency overlaps the WMMA stream.
    v16bf a = make_frag(xrow + k0 + kA, xrow + k0 + kA + 16);
    v16bf b[8];
#pragma unroll
    for (int t = 0; t < 8; ++t) {
      const u16* bp = w1base + (size_t)t * 16 * D_ + k0;
      b[t] = make_frag(bp, bp + 16);
    }
#pragma unroll
    for (int t = 0; t < 8; ++t)
      acc[t] = __builtin_amdgcn_wmma_f32_16x16x32_bf16(
          false, a, false, b[t], (short)0, acc[t], false, false);
  }

  // C/D layout: lanes 0-15: (M=r, N=l16); lanes 16-31: (M=8+r, N=l16)
  const int cm0 = half ? 8 : 0;
#pragma unroll
  for (int t = 0; t < 8; ++t) {
    const int n = n_base + t * 16 + l16;
    const float bias = b1[e * H_ + n];
#pragma unroll
    for (int r = 0; r < 8; ++r) {
      float v = acc[t][r] + bias;
      v = v > 0.f ? v : 0.f;   // ReLU
      s_h[(cm0 + r) * H_ + n] = f32_to_bf16(v);
    }
  }
  __syncthreads();

  // ---------------- layer 2 ----------------
  const u16* hrow = s_h + l16 * H_;
  const int n2 = wave * 32;  // 16 waves cover O_ = 512
  const u16* w2base = W2t + ((size_t)e * O_ + n2 + l16) * H_ + kB;
  v8f acc2[2];
  acc2[0] = vzero; acc2[1] = vzero;

  for (int k0 = 0; k0 < H_; k0 += 32) {
    v16bf a = make_frag(hrow + k0 + kA, hrow + k0 + kA + 16);
    v16bf b[2];
#pragma unroll
    for (int t = 0; t < 2; ++t) {
      const u16* bp = w2base + (size_t)t * 16 * H_ + k0;
      b[t] = make_frag(bp, bp + 16);
    }
#pragma unroll
    for (int t = 0; t < 2; ++t)
      acc2[t] = __builtin_amdgcn_wmma_f32_16x16x32_bf16(
          false, a, false, b[t], (short)0, acc2[t], false, false);
  }

  // combined[row] += gate * exp(out + b2); padded rows (idx >= cnt) skipped.
#pragma unroll
  for (int t = 0; t < 2; ++t) {
    const int o = n2 + t * 16 + l16;
    const float bias = b2[e * O_ + o];
#pragma unroll
    for (int r = 0; r < 8; ++r) {
      const int m   = cm0 + r;
      const int idx = tile * 16 + m;
      if (idx < cnt) {
        const float g = gate_list[(size_t)e * B_ + idx];
        const int row = rows_list[(size_t)e * B_ + idx];
        atomicAdd(&combined[(size_t)row * O_ + o], g * __expf(acc2[t][r] + bias));
      }
    }
  }
}

// ---------------------------------------------------------------------------
// y = log(combined == 0 ? EPS : combined)
// ---------------------------------------------------------------------------
__global__ __launch_bounds__(256) void finalize_kernel(
    const float* __restrict__ combined, float* __restrict__ y) {
  const size_t i = (size_t)blockIdx.x * 256 + threadIdx.x;
  float c = combined[i];
  const float EPSF = 2.2204460492503131e-16f;  // float64 eps, per reference
  y[i] = __logf(c == 0.f ? EPSF : c);
}

// ---------------------------------------------------------------------------
extern "C" void kernel_launch(void* const* d_in, const int* in_sizes, int n_in,
                              void* d_out, int out_size, void* d_ws, size_t ws_size,
                              hipStream_t stream) {
  (void)in_sizes; (void)n_in; (void)out_size; (void)ws_size;
  const float* x      = (const float*)d_in[0];
  // d_in[1] = Cats (unused by reference)
  const float* w_gate = (const float*)d_in[2];
  const float* W1     = (const float*)d_in[3];
  const float* b1     = (const float*)d_in[4];
  const float* W2     = (const float*)d_in[5];
  const float* b2     = (const float*)d_in[6];
  // d_in[7] = train (unused)

  char* ws = (char*)d_ws;
  size_t off = 0;
  auto carve = [&](size_t bytes) -> void* {
    void* p = ws + off;
    off += (bytes + 255) & ~(size_t)255;
    return p;
  };
  float* combined   = (float*)carve((size_t)B_ * O_ * 4);        // 16.8 MB
  u16*   xbf        = (u16*)  carve((size_t)B_ * D_ * 2);        // 50.3 MB
  u16*   W1t        = (u16*)  carve((size_t)E_ * H_ * D_ * 2);   // 100.7 MB
  u16*   W2t        = (u16*)  carve((size_t)E_ * O_ * H_ * 2);   // 16.8 MB
  int*   rows_list  = (int*)  carve((size_t)E_ * B_ * 4);
  float* gate_list  = (float*)carve((size_t)E_ * B_ * 4);
  int*   counts     = (int*)  carve(64);
  float* importance = (float*)carve(64);

  (void)hipMemsetAsync(combined, 0, (size_t)B_ * O_ * 4, stream);
  (void)hipMemsetAsync(counts, 0, 64, stream);
  (void)hipMemsetAsync(importance, 0, 64, stream);

  transpose_to_bf16<<<dim3(H_ / 32, D_ / 32, E_), dim3(32, 8), 0, stream>>>(W1, W1t, D_, H_);
  transpose_to_bf16<<<dim3(O_ / 32, H_ / 32, E_), dim3(32, 8), 0, stream>>>(W2, W2t, H_, O_);
  gate_kernel<<<B_, 256, 0, stream>>>(x, w_gate, xbf, counts, importance,
                                      rows_list, gate_list);
  aux_kernel<<<1, 1, 0, stream>>>(importance, counts, (float*)d_out + (size_t)B_ * O_);
  ffn_kernel<<<dim3(512, E_), 512, 0, stream>>>(xbf, W1t, b1, W2t, b2, counts,
                                                rows_list, gate_list, combined);
  finalize_kernel<<<(B_ * O_) / 256, 256, 0, stream>>>(combined, (float*)d_out);
}
